// ECNetWrapperGNN_26620207301025
// MI455X (gfx1250) — matrix operand order repaired and checked by reading, not compiled, not run on hardware
//
#include <hip/hip_runtime.h>
#include <hip/hip_bf16.h>
#include <stddef.h>
#include <stdint.h>

typedef __bf16 bf16_t;
typedef __attribute__((ext_vector_type(16))) __bf16 v16bf;
typedef __attribute__((ext_vector_type(8)))  __bf16 v8bf;
typedef __attribute__((ext_vector_type(8)))  float  v8f;

static constexpr int kN  = 50000;   // nodes  (multiple of 16)
static constexpr int kE  = 800000;  // edges (before self loops)
static constexpr int kB  = 512;     // graphs (multiple of 16)
static constexpr int kH  = 4;       // heads
static constexpr int kC  = 16;      // channels per head
static constexpr int kF  = 64;      // H*C
static constexpr float kSlope = 0.2f;
static_assert(kN % 16 == 0 && kB % 16 == 0, "full WMMA tiles assumed");

// ---------------- utility kernels ----------------

__global__ void fill_f32(float* p, float v, int n) {
  int i = blockIdx.x * blockDim.x + threadIdx.x;
  if (i < n) p[i] = v;
}

// x = concat(node_feats, global[batch]) zero-padded to 64, converted to bf16
__global__ void prep_x(const float* __restrict__ nf, const float* __restrict__ gf,
                       const int* __restrict__ batch, bf16_t* __restrict__ xA) {
  int i = blockIdx.x * blockDim.x + threadIdx.x;
  if (i >= kN * kF) return;
  int n = i >> 6, f = i & 63;
  float v = 0.f;
  if (f < 9)       v = nf[n * 9 + f];
  else if (f < 13) v = gf[batch[n] * 4 + (f - 9)];
  xA[i] = (bf16_t)v;
}

// Repack weight (rows x 64, rows<=64, f32) into bf16 WMMA B-fragment order:
// Wfrag[((ks*4+t)*32 + lane)*16 + e] = W[(ks*32 + (lane>>4)*16 + e) * 64 + t*16 + (lane&15)]
__global__ void pack_w(const float* __restrict__ W, int rows, bf16_t* __restrict__ Wfrag) {
  int i = blockIdx.x * blockDim.x + threadIdx.x;
  if (i >= 8 * 32 * 16) return;
  int e    = i & 15;
  int lane = (i >> 4) & 31;
  int fi   = i >> 9;            // 0..7 = ks*4 + t
  int ks   = fi >> 2, t = fi & 3;
  int k = ks * 32 + (lane >> 4) * 16 + e;
  int n = t * 16 + (lane & 15);
  Wfrag[i] = (bf16_t)((k < rows) ? W[k * 64 + n] : 0.f);
}

// ---------------- WMMA GEMM: C[M x 64] = A[M x 64] * W[64 x 64] ----------------
// W staged to LDS via async global->LDS DMA (ASYNCcnt); 8 B fragments resident
// in VGPRs; each wave handles 4 full 16-row M tiles (32 static v_wmma).
__global__ void gemm_bf16_64(const bf16_t* __restrict__ A, const bf16_t* __restrict__ Wfrag,
                             float* __restrict__ Cm, int M) {
  __shared__ bf16_t lW[8 * 32 * 16];     // 8 KB, fragment-ordered
  int tid = threadIdx.x;

  // async stage: 128 threads x 4 chunks x 16B = 8KB
  unsigned ldsbase = (unsigned)(uintptr_t)(&lW[0]);
  const char* gbase = (const char*)Wfrag;
  #pragma unroll
  for (int j = 0; j < 4; j++) {
    unsigned off = (unsigned)(tid + j * 128) * 16u;
    asm volatile("global_load_async_to_lds_b128 %0, %1, off"
                 :: "v"(ldsbase + off), "v"(gbase + off)
                 : "memory");
  }
  asm volatile("s_wait_asynccnt 0x0" ::: "memory");
  __syncthreads();

  int lane = tid & 31;
  int wv   = tid >> 5;
  int mrow  = lane & 15;                 // A: M index within tile
  int khalf = (lane >> 4) * 16;          // A: half-wave K split
  int ncol  = lane & 15;                 // C: N index
  int mhi   = (lane >> 4) * 8;           // C: M split

  // B fragments from LDS (ds_load_b128 x2 each)
  v16bf bfrag[8];
  #pragma unroll
  for (int fi = 0; fi < 8; fi++) {
    const bf16_t* bp = lW + ((size_t)(fi * 32 + lane)) * 16;
    v8bf lo = *(const v8bf*)(bp);
    v8bf hi = *(const v8bf*)(bp + 8);
    #pragma unroll
    for (int e = 0; e < 8; e++) { bfrag[fi][e] = lo[e]; bfrag[fi][e + 8] = hi[e]; }
  }

  int tile0 = (blockIdx.x * (blockDim.x >> 5) + wv) * 4;
  #pragma unroll
  for (int tt = 0; tt < 4; tt++) {
    int mbase = (tile0 + tt) * 16;
    if (mbase < M) {                     // wave-uniform: EXEC all-ones at WMMA
      const bf16_t* arow = A + (size_t)(mbase + mrow) * kF;
      __builtin_prefetch(arow + kF, 0, 1);

      v8f acc[4];
      #pragma unroll
      for (int t = 0; t < 4; t++)
        #pragma unroll
        for (int r = 0; r < 8; r++) acc[t][r] = 0.f;

      #pragma unroll
      for (int ks = 0; ks < 2; ks++) {
        int kbase = ks * 32;
        v8bf lo = *(const v8bf*)(arow + kbase + khalf);
        v8bf hi = *(const v8bf*)(arow + kbase + khalf + 8);
        v16bf a;
        #pragma unroll
        for (int e = 0; e < 8; e++) { a[e] = lo[e]; a[e + 8] = hi[e]; }
        #pragma unroll
        for (int t = 0; t < 4; t++)
          acc[t] = __builtin_amdgcn_wmma_f32_16x16x32_bf16(
              false, a, false, bfrag[ks * 4 + t], (short)0, acc[t], false, false);
      }

      // C/D layout: VGPR r, lanes0-15 -> M=r; lanes16-31 -> M=r+8; N = lane&15
      float* crow = Cm + (size_t)(mbase + mhi) * kF + ncol;
      #pragma unroll
      for (int t = 0; t < 4; t++)
        #pragma unroll
        for (int r = 0; r < 8; r++)
          crow[(size_t)r * kF + t * 16] = acc[t][r];
    }
  }
}

// ---------------- GAT attention ----------------

// per-node: alpha_src/alpha_dst for all 4 heads from one 64-float read
__global__ void alpha_k(const float* __restrict__ h, const float* __restrict__ asr,
                        const float* __restrict__ adt, float* __restrict__ aS,
                        float* __restrict__ aD) {
  int n = blockIdx.x * blockDim.x + threadIdx.x;
  if (n >= kN) return;
  const float4* hp = (const float4*)(h + (size_t)n * kF);
  float4* oS = (float4*)(aS + (size_t)n * kH);
  float4* oD = (float4*)(aD + (size_t)n * kH);
  float s[kH], d[kH];
  #pragma unroll
  for (int hh = 0; hh < kH; hh++) { s[hh] = 0.f; d[hh] = 0.f; }
  #pragma unroll
  for (int j = 0; j < 16; j++) {
    float4 hv = hp[j];
    int hh = j >> 2, c0 = (j & 3) * 4;
    const float* ap = asr + hh * kC + c0;
    const float* dp = adt + hh * kC + c0;
    s[hh] += hv.x * ap[0] + hv.y * ap[1] + hv.z * ap[2] + hv.w * ap[3];
    d[hh] += hv.x * dp[0] + hv.y * dp[1] + hv.z * dp[2] + hv.w * dp[3];
  }
  *oS = make_float4(s[0], s[1], s[2], s[3]);
  *oD = make_float4(d[0], d[1], d[2], d[3]);
}

__device__ inline void edge_sd(int e, int& src, int& dst, const int* ei) {
  if (e < kE) { src = ei[e]; dst = ei[kE + e]; }
  else        { src = dst = e - kE; }
}

__device__ inline float lrelu(float v) { return v > 0.f ? v : kSlope * v; }

__device__ inline void atomicMaxF(float* addr, float val) {
  int* ai = (int*)addr;
  int old = *ai;
  while (true) {
    float of = __int_as_float(old);
    if (of >= val) break;
    int assumed = old;
    old = atomicCAS(ai, assumed, __float_as_int(val));
    if (old == assumed) break;
  }
}

__global__ void edge_max(const int* __restrict__ ei, const float* __restrict__ aS,
                         const float* __restrict__ aD, float* __restrict__ mM) {
  int e = blockIdx.x * blockDim.x + threadIdx.x;
  if (e >= kE + kN) return;
  int src, dst; edge_sd(e, src, dst, ei);
  float4 s4 = *(const float4*)(aS + (size_t)src * kH);
  float4 d4 = *(const float4*)(aD + (size_t)dst * kH);
  float* mp = mM + (size_t)dst * kH;
  atomicMaxF(mp + 0, lrelu(s4.x + d4.x));
  atomicMaxF(mp + 1, lrelu(s4.y + d4.y));
  atomicMaxF(mp + 2, lrelu(s4.z + d4.z));
  atomicMaxF(mp + 3, lrelu(s4.w + d4.w));
}

__global__ void edge_sum(const int* __restrict__ ei, const float* __restrict__ aS,
                         const float* __restrict__ aD, const float* __restrict__ mM,
                         float* __restrict__ sS) {
  int e = blockIdx.x * blockDim.x + threadIdx.x;
  if (e >= kE + kN) return;
  int src, dst; edge_sd(e, src, dst, ei);
  float4 s4 = *(const float4*)(aS + (size_t)src * kH);
  float4 d4 = *(const float4*)(aD + (size_t)dst * kH);
  float4 m4 = *(const float4*)(mM + (size_t)dst * kH);
  float* sp = sS + (size_t)dst * kH;
  atomicAdd(sp + 0, __expf(lrelu(s4.x + d4.x) - m4.x));
  atomicAdd(sp + 1, __expf(lrelu(s4.y + d4.y) - m4.y));
  atomicAdd(sp + 2, __expf(lrelu(s4.z + d4.z) - m4.z));
  atomicAdd(sp + 3, __expf(lrelu(s4.w + d4.w) - m4.w));
}

__global__ void edge_agg(const int* __restrict__ ei, const float* __restrict__ aS,
                         const float* __restrict__ aD, const float* __restrict__ mM,
                         const float* __restrict__ sS, const float* __restrict__ h,
                         float* __restrict__ outF) {
  int e = blockIdx.x * blockDim.x + threadIdx.x;
  if (e >= kE + kN) return;
  int src, dst; edge_sd(e, src, dst, ei);
  float4 s4 = *(const float4*)(aS + (size_t)src * kH);
  float4 d4 = *(const float4*)(aD + (size_t)dst * kH);
  float4 m4 = *(const float4*)(mM + (size_t)dst * kH);
  float4 z4 = *(const float4*)(sS + (size_t)dst * kH);
  float w[kH];
  w[0] = __expf(lrelu(s4.x + d4.x) - m4.x) / z4.x;
  w[1] = __expf(lrelu(s4.y + d4.y) - m4.y) / z4.y;
  w[2] = __expf(lrelu(s4.z + d4.z) - m4.z) / z4.z;
  w[3] = __expf(lrelu(s4.w + d4.w) - m4.w) / z4.w;
  const float4* hp = (const float4*)(h + (size_t)src * kF);
  float* op = outF + (size_t)dst * kF;
  #pragma unroll
  for (int j = 0; j < 16; j++) {
    float4 hv = hp[j];
    float ww = w[j >> 2];
    atomicAdd(op + j * 4 + 0, hv.x * ww);
    atomicAdd(op + j * 4 + 1, hv.y * ww);
    atomicAdd(op + j * 4 + 2, hv.z * ww);
    atomicAdd(op + j * 4 + 3, hv.w * ww);
  }
}

// bias + optional relu; write back f32 in place and bf16 for next layer's GEMM
__global__ void post_k(float* __restrict__ outF, const float* __restrict__ bias,
                       bf16_t* __restrict__ xA, int do_relu) {
  int i = blockIdx.x * blockDim.x + threadIdx.x;
  if (i >= kN * kF) return;
  int f = i & 63;
  float v = outF[i] + bias[f];
  if (do_relu) v = fmaxf(v, 0.f);
  outF[i] = v;
  xA[i] = (bf16_t)v;
}

// ---------------- pooling + MLP ----------------

__global__ void pool_sum(const int* __restrict__ batch, const float* __restrict__ x,
                         float* __restrict__ pS) {
  int i = blockIdx.x * blockDim.x + threadIdx.x;
  if (i >= kN * kF) return;
  int n = i >> 6, f = i & 63;
  atomicAdd(&pS[batch[n] * kF + f], x[i]);
}

__global__ void pool_cnt(const int* __restrict__ batch, float* __restrict__ cnt) {
  int i = blockIdx.x * blockDim.x + threadIdx.x;
  if (i >= kN) return;
  atomicAdd(&cnt[batch[i]], 1.f);
}

__global__ void pool_fin(float* __restrict__ pS, const float* __restrict__ cnt,
                         bf16_t* __restrict__ mX) {
  int i = blockIdx.x * blockDim.x + threadIdx.x;
  if (i >= kB * kF) return;
  int b = i >> 6;
  float c = cnt[b]; if (c < 1.f) c = 1.f;
  float v = pS[i] / c;
  pS[i] = v;
  mX[i] = (bf16_t)v;
}

__global__ void mlp_post(float* __restrict__ hF, const float* __restrict__ bias,
                         bf16_t* __restrict__ mX) {
  int i = blockIdx.x * blockDim.x + threadIdx.x;
  if (i >= kB * kF) return;
  int f = i & 63;
  float v = fmaxf(hF[i] + bias[f], 0.f);
  hF[i] = v;
  mX[i] = (bf16_t)v;
}

__global__ void final_k(const float* __restrict__ hF, const float* __restrict__ W3,
                        const float* __restrict__ b3, float* __restrict__ out) {
  int b = blockIdx.x * blockDim.x + threadIdx.x;
  if (b >= kB) return;
  float s = b3[0];
  #pragma unroll
  for (int f = 0; f < kF; f++) s += hF[(size_t)b * kF + f] * W3[f];
  out[b] = s;
}

// ---------------- launch ----------------

extern "C" void kernel_launch(void* const* d_in, const int* in_sizes, int n_in,
                              void* d_out, int out_size, void* d_ws, size_t ws_size,
                              hipStream_t stream) {
  const float* nf    = (const float*)d_in[0];
  const float* gf    = (const float*)d_in[1];
  const int*   ei    = (const int*)d_in[2];
  const int*   batch = (const int*)d_in[3];
  const float* Wl[3]  = {(const float*)d_in[4],  (const float*)d_in[8],  (const float*)d_in[12]};
  const float* asr[3] = {(const float*)d_in[5],  (const float*)d_in[9],  (const float*)d_in[13]};
  const float* adt[3] = {(const float*)d_in[6],  (const float*)d_in[10], (const float*)d_in[14]};
  const float* bl[3]  = {(const float*)d_in[7],  (const float*)d_in[11], (const float*)d_in[15]};
  const float* mW[4]  = {(const float*)d_in[16], (const float*)d_in[18], (const float*)d_in[20], (const float*)d_in[22]};
  const float* mBi[4] = {(const float*)d_in[17], (const float*)d_in[19], (const float*)d_in[21], (const float*)d_in[23]};
  float* out = (float*)d_out;

  char* ws = (char*)d_ws;
  size_t off = 0;
  auto alloc = [&](size_t bytes) -> char* {
    char* p = ws + off; off += (bytes + 255) & ~(size_t)255; return p;
  };
  bf16_t* xA = (bf16_t*)alloc((size_t)kN * kF * 2);
  float*  hF = (float*) alloc((size_t)kN * kF * 4);
  float*  oF = (float*) alloc((size_t)kN * kF * 4);
  float*  aS = (float*) alloc((size_t)kN * kH * 4);
  float*  aD = (float*) alloc((size_t)kN * kH * 4);
  float*  mM = (float*) alloc((size_t)kN * kH * 4);
  float*  sS = (float*) alloc((size_t)kN * kH * 4);
  bf16_t* Wf = (bf16_t*)alloc(8 * 32 * 16 * 2);     // fragment-ordered weights
  float*  pS = (float*) alloc((size_t)kB * kF * 4);
  float*  cnt= (float*) alloc((size_t)kB * 4);
  float*  mF = (float*) alloc((size_t)kB * kF * 4);
  bf16_t* mX = (bf16_t*)alloc((size_t)kB * kF * 2);

  const int T = 256;
  auto g = [](long n, int t) { return (int)((n + t - 1) / t); };
  auto gemm_blocks = [](int M) {            // 4 waves x 4 tiles = 16 tiles/block
    int tiles = (M + 15) / 16;
    int waves = (tiles + 3) / 4;
    return (waves + 3) / 4;
  };

  prep_x<<<g((long)kN * kF, T), T, 0, stream>>>(nf, gf, batch, xA);

  const long nEdge = kE + kN;
  for (int l = 0; l < 3; l++) {
    pack_w<<<g(4096, T), T, 0, stream>>>(Wl[l], l == 0 ? 13 : 64, Wf);
    gemm_bf16_64<<<gemm_blocks(kN), 128, 0, stream>>>(xA, Wf, hF, kN);
    alpha_k<<<g(kN, T), T, 0, stream>>>(hF, asr[l], adt[l], aS, aD);
    fill_f32<<<g((long)kN * kH, T), T, 0, stream>>>(mM, -1e30f, kN * kH);
    fill_f32<<<g((long)kN * kH, T), T, 0, stream>>>(sS, 0.f, kN * kH);
    fill_f32<<<g((long)kN * kF, T), T, 0, stream>>>(oF, 0.f, kN * kF);
    edge_max<<<g(nEdge, T), T, 0, stream>>>(ei, aS, aD, mM);
    edge_sum<<<g(nEdge, T), T, 0, stream>>>(ei, aS, aD, mM, sS);
    edge_agg<<<g(nEdge, T), T, 0, stream>>>(ei, aS, aD, mM, sS, hF, oF);
    post_k<<<g((long)kN * kF, T), T, 0, stream>>>(oF, bl[l], xA, l < 2 ? 1 : 0);
  }

  // scatter-mean pooling
  fill_f32<<<g((long)kB * kF, T), T, 0, stream>>>(pS, 0.f, kB * kF);
  fill_f32<<<g(kB, T), T, 0, stream>>>(cnt, 0.f, kB);
  pool_sum<<<g((long)kN * kF, T), T, 0, stream>>>(batch, oF, pS);
  pool_cnt<<<g(kN, T), T, 0, stream>>>(batch, cnt);
  pool_fin<<<g((long)kB * kF, T), T, 0, stream>>>(pS, cnt, mX);

  // MLP: 3x (64x64 GEMM + bias + relu), then 64->1
  for (int j = 0; j < 3; j++) {
    pack_w<<<g(4096, T), T, 0, stream>>>(mW[j], 64, Wf);
    gemm_bf16_64<<<gemm_blocks(kB), 128, 0, stream>>>(mX, Wf, mF, kB);
    mlp_post<<<g((long)kB * kF, T), T, 0, stream>>>(mF, mBi[j], mX);
  }
  final_k<<<g(kB, T), T, 0, stream>>>(mF, mW[3], mBi[3], out);
}
